// GeneralModelMamba_21663815041678
// MI455X (gfx1250) — compile-verified, matrix operand order checked by reading, not run on hardware
//
#include <hip/hip_runtime.h>
#include <hip/hip_bf16.h>
#include <math.h>

// ---------------- model constants ----------------
#define D_MODEL   192
#define D_STATE   64
#define HEADDIM   48
#define D_CONV    4
#define NLAYERS   8
#define D_INNER   384           // EXPAND * D_MODEL
#define NHEADS    8             // D_INNER / HEADDIM
#define CONV_DIM  512           // D_INNER + 2*D_STATE
#define D_IN_PROJ 904           // 2*D_INNER + 2*D_STATE + NHEADS
#define NPAD      960           // D_IN_PROJ padded to /64 (N-strip width)
#define BATCH     4
#define SEQLEN    4096
#define MTOT      (BATCH * SEQLEN)   // 16384 token rows
#define OUTPUT_DIM 4
#define INPUT_DIM  3
#define KMAX      384

typedef unsigned short u16;
typedef __attribute__((ext_vector_type(16))) __bf16 v16bf;
typedef __attribute__((ext_vector_type(8)))  __bf16 v8bf;
typedef __attribute__((ext_vector_type(8)))  float  v8f;

// Tensor Data Mover (this toolchain: clang-23 6-arg builtin form)
#if defined(__has_builtin)
#if __has_builtin(__builtin_amdgcn_tensor_load_to_lds) && __has_builtin(__builtin_amdgcn_s_wait_tensorcnt)
#define HAS_TDM 1
#endif
#endif
#ifndef HAS_TDM
#define HAS_TDM 0
#endif

#if HAS_TDM
typedef __attribute__((ext_vector_type(4))) unsigned int u32x4;
typedef __attribute__((ext_vector_type(8))) int         i32x8;
typedef __attribute__((ext_vector_type(4))) int         i32x4;
#endif

// float -> bf16 bits, round-to-nearest-even
__device__ __forceinline__ u16 f2bf(float f) {
    union { float f; unsigned u; } v; v.f = f;
    unsigned r = v.u + 0x7FFFu + ((v.u >> 16) & 1u);
    return (u16)(r >> 16);
}
__device__ __forceinline__ float sigmoidf_(float x) {
    return 1.f / (1.f + __expf(-x));
}

// ---------------- weight packing (f32 -> bf16, pad in_proj N to 960) -------
__global__ void k_pack_inw(const float* __restrict__ w, u16* __restrict__ wb) {
    int i = blockIdx.x * blockDim.x + threadIdx.x;
    if (i >= NLAYERS * NPAD * D_MODEL) return;
    int layer = i / (NPAD * D_MODEL);
    int rem   = i % (NPAD * D_MODEL);
    int r     = rem / D_MODEL;
    int k     = rem % D_MODEL;
    float v = (r < D_IN_PROJ) ? w[(layer * D_IN_PROJ + r) * D_MODEL + k] : 0.f;
    wb[i] = f2bf(v);
}
__global__ void k_pack_outw(const float* __restrict__ w, u16* __restrict__ wb) {
    int i = blockIdx.x * blockDim.x + threadIdx.x;
    if (i >= NLAYERS * D_MODEL * D_INNER) return;
    wb[i] = f2bf(w[i]);
}

// ---------------- input linear + relu ----------------
// inp: (B, 3, L) ; x,xb: (M, 192) with m = b*L + l
__global__ void k_lin_in(const float* __restrict__ inp, const float* __restrict__ w,
                         const float* __restrict__ bias,
                         float* __restrict__ x, u16* __restrict__ xb) {
    int i = blockIdx.x * blockDim.x + threadIdx.x;
    if (i >= MTOT * D_MODEL) return;
    int m = i / D_MODEL, d = i % D_MODEL;
    int b = m / SEQLEN,  l = m % SEQLEN;
    float acc = bias[d];
#pragma unroll
    for (int k = 0; k < INPUT_DIM; ++k)
        acc += inp[(b * INPUT_DIM + k) * SEQLEN + l] * w[d * INPUT_DIM + k];
    acc = fmaxf(acc, 0.f);
    x[i]  = acc;
    xb[i] = f2bf(acc);
}

// ---------------- WMMA bf16 GEMM: C[M x ldc] = A[M x K] * B[N x K]^T --------
// Block = 8 waves = 128(M) x 64(N) C tile. The 64-row B strip (64 x K bf16)
// is staged into LDS once per block by the Tensor Data Mover (wave 0 issues
// tensor_load_to_lds, waits TENSORcnt, then the workgroup barrier publishes
// LDS). Each wave owns a 16(M) x 64(N) strip: one A fragment feeds 4 WMMAs.
// Grids divide exactly (M=16384 = 128*128; N in {960,192} = 64*grid.y), so
// EXEC is all-ones for every WMMA as the ISA requires.
// ISA 16-bit fragment layout: lanes 0-15 hold row m, K = {0..7, 16..23};
// lanes 16-31 hold row m, K = {8..15, 24..31}; B mirrored (lane n = column n
// of B == row n of the NxK weight array).
#define GEMM_WAVES 8
__global__ void k_gemm_bf16(const u16* __restrict__ A, const u16* __restrict__ B,
                            float* __restrict__ C, int M, int N, int K, int ldc) {
    __shared__ u16 Bs[64 * KMAX];
    const int lane = threadIdx.x & 31;
    const int wave = threadIdx.x >> 5;
    const int m0   = (blockIdx.x * GEMM_WAVES + wave) * 16;
    const int n0   = blockIdx.y * 64;

#if HAS_TDM
    if (wave == 0) {
        const unsigned lds_off = (unsigned)(size_t)(&Bs[0]);
        const unsigned long long ga =
            (unsigned long long)(size_t)(B + (size_t)n0 * K);
        u32x4 g0;
        g0[0] = 1u;                                   // count=1, load, no gather
        g0[1] = lds_off;                              // lds_addr (bytes)
        g0[2] = (unsigned)ga;                         // global_addr[31:0]
        g0[3] = (unsigned)((ga >> 32) & 0x01FFFFFFu)  // global_addr[56:32]
                | 0x80000000u;                        // type=2 ("image")
        i32x8 g1;
        g1[0] = (int)(1u << 16);                      // data_size=1 (2 bytes)
        g1[1] = (int)((unsigned)(K & 0xFFFF) << 16);  // tensor_dim0[15:0]
        g1[2] = (int)(((unsigned)(K >> 16) & 0xFFFFu) |
                      ((unsigned)(N & 0xFFFF) << 16));// tdim0 hi | tensor_dim1 lo
        g1[3] = (int)(((unsigned)(N >> 16) & 0xFFFFu) |
                      ((unsigned)(K & 0xFFFF) << 16));// tdim1 hi | tile_dim0=K
        g1[4] = 64;                                   // tile_dim1=64 rows, tile_dim2=0
        g1[5] = K;                                    // tensor_dim0_stride[31:0]
        g1[6] = 0;                                    // stride0 hi | stride1 lo
        g1[7] = 0;                                    // stride1 hi
        i32x4 zz4 = {0, 0, 0, 0};
        i32x8 zz8 = {0, 0, 0, 0, 0, 0, 0, 0};
        __builtin_amdgcn_tensor_load_to_lds(g0, g1, zz4, zz4, zz8, 0);
        __builtin_amdgcn_s_wait_tensorcnt(0);
    }
#else
    for (int i = threadIdx.x; i < 64 * K; i += 32 * GEMM_WAVES)
        Bs[i] = B[(size_t)n0 * K + i];
#endif
    __syncthreads();

    const int half = lane >> 4;               // 0: lanes 0-15, 1: lanes 16-31
    const int idx  = lane & 15;               // A row / B col within tile
    const u16* arow = A + (size_t)(m0 + idx) * K;
    const u16* brow = Bs + idx * K;           // + nt*16*K per N sub-tile
    v8f acc0 = {}, acc1 = {}, acc2 = {}, acc3 = {};
    for (int k0 = 0; k0 < K; k0 += 32) {
        const int kb = k0 + half * 8;
        __builtin_prefetch(arow + kb + 64, 0, 0);   // global_prefetch_b8
        v16bf av, bv0, bv1, bv2, bv3;
        ((v8bf*)&av)[0]  = *(const v8bf*)(arow + kb);
        ((v8bf*)&av)[1]  = *(const v8bf*)(arow + kb + 16);
        ((v8bf*)&bv0)[0] = *(const v8bf*)(brow + kb);
        ((v8bf*)&bv0)[1] = *(const v8bf*)(brow + kb + 16);
        ((v8bf*)&bv1)[0] = *(const v8bf*)(brow + 16 * K + kb);
        ((v8bf*)&bv1)[1] = *(const v8bf*)(brow + 16 * K + kb + 16);
        ((v8bf*)&bv2)[0] = *(const v8bf*)(brow + 32 * K + kb);
        ((v8bf*)&bv2)[1] = *(const v8bf*)(brow + 32 * K + kb + 16);
        ((v8bf*)&bv3)[0] = *(const v8bf*)(brow + 48 * K + kb);
        ((v8bf*)&bv3)[1] = *(const v8bf*)(brow + 48 * K + kb + 16);
        acc0 = __builtin_amdgcn_wmma_f32_16x16x32_bf16(false, av, false, bv0, (short)0, acc0, false, false);
        acc1 = __builtin_amdgcn_wmma_f32_16x16x32_bf16(false, av, false, bv1, (short)0, acc1, false, false);
        acc2 = __builtin_amdgcn_wmma_f32_16x16x32_bf16(false, av, false, bv2, (short)0, acc2, false, false);
        acc3 = __builtin_amdgcn_wmma_f32_16x16x32_bf16(false, av, false, bv3, (short)0, acc3, false, false);
    }
    // C/D layout: VGPR r -> row m0+r (lanes 0-15) / m0+8+r (lanes 16-31)
    float* cbase = C + (size_t)(m0 + half * 8) * ldc + n0 + idx;
#pragma unroll
    for (int r = 0; r < 8; ++r) {
        float* cr = cbase + (size_t)r * ldc;
        cr[0]  = acc0[r];
        cr[16] = acc1[r];
        cr[32] = acc2[r];
        cr[48] = acc3[r];
    }
    (void)M;
}

// ---------------- dt = softplus(raw + dt_bias) ----------------
__global__ void k_dt(const float* __restrict__ zxb, const float* __restrict__ dt_bias,
                     float* __restrict__ dt) {
    int i = blockIdx.x * blockDim.x + threadIdx.x;
    if (i >= MTOT * NHEADS) return;
    int m = i / NHEADS, h = i % NHEADS;
    float v = zxb[(size_t)m * NPAD + (D_INNER + CONV_DIM) + h] + dt_bias[h];
    dt[i] = (v > 20.f) ? v : log1pf(__expf(v));
}

// ---------------- causal depthwise conv (width 4) + SiLU ----------------
__global__ void k_conv(const float* __restrict__ zxb, const float* __restrict__ cw,
                       const float* __restrict__ cb, float* __restrict__ xBC) {
    int i = blockIdx.x * blockDim.x + threadIdx.x;
    if (i >= MTOT * CONV_DIM) return;
    int m = i / CONV_DIM, c = i % CONV_DIM;
    int l = m & (SEQLEN - 1);
    float acc = cb[c];
#pragma unroll
    for (int k = 0; k < D_CONV; ++k) {
        int ll = l - (D_CONV - 1) + k;
        if (ll >= 0)
            acc += zxb[(size_t)(m - (D_CONV - 1) + k) * NPAD + D_INNER + c] * cw[c * D_CONV + k];
    }
    xBC[i] = acc * sigmoidf_(acc);
}

// ---------------- sequential selective-state scan ----------------
// One block per (batch, head). 8 waves: wave w owns head-dim rows p = 6w..6w+5;
// lane owns state cols n = lane, lane+32. y = sum_n h*C via shfl_xor reduce.
__global__ void k_scan(const float* __restrict__ xBC, const float* __restrict__ dt,
                       const float* __restrict__ A_log, float* __restrict__ ys) {
    const int b    = blockIdx.x;
    const int h    = blockIdx.y;
    const int lane = threadIdx.x & 31;
    const int wave = threadIdx.x >> 5;
    const float Ah = -__expf(A_log[h]);
    float hs0[6], hs1[6];
#pragma unroll
    for (int j = 0; j < 6; ++j) { hs0[j] = 0.f; hs1[j] = 0.f; }
    const int p0 = wave * 6;
    for (int l = 0; l < SEQLEN; ++l) {
        const int m = b * SEQLEN + l;
        const float dtv = dt[m * NHEADS + h];
        const float dA  = __expf(dtv * Ah);
        const size_t base = (size_t)m * CONV_DIM;
        const float Bn0 = xBC[base + D_INNER + lane];
        const float Bn1 = xBC[base + D_INNER + 32 + lane];
        const float Cn0 = xBC[base + D_INNER + D_STATE + lane];
        const float Cn1 = xBC[base + D_INNER + D_STATE + 32 + lane];
#pragma unroll
        for (int j = 0; j < 6; ++j) {
            const int p = p0 + j;
            const float xv = xBC[base + h * HEADDIM + p];
            const float cf = dtv * xv;
            hs0[j] = hs0[j] * dA + cf * Bn0;
            hs1[j] = hs1[j] * dA + cf * Bn1;
            float y = hs0[j] * Cn0 + hs1[j] * Cn1;
#pragma unroll
            for (int off = 16; off; off >>= 1) y += __shfl_xor(y, off, 32);
            if (lane == 0) ys[(size_t)m * D_INNER + h * HEADDIM + p] = y;
        }
    }
}

// ---------------- y = (ys + D*xh) * silu(z); RMS-norm; -> bf16 ------------
__global__ void k_gate_rms(const float* __restrict__ ys, const float* __restrict__ xBC,
                           const float* __restrict__ zxb, const float* __restrict__ Dv,
                           const float* __restrict__ norm_w, u16* __restrict__ yb) {
    __shared__ float red[128];
    const int m = blockIdx.x;
    const int t = threadIdx.x;     // 0..127, 3 elements each
    float g[3];
    float ss = 0.f;
#pragma unroll
    for (int j = 0; j < 3; ++j) {
        int d = t + j * 128;
        int h = d / HEADDIM;
        float xv = xBC[(size_t)m * CONV_DIM + d];
        float yv = ys[(size_t)m * D_INNER + d] + Dv[h] * xv;
        float zv = zxb[(size_t)m * NPAD + d];
        g[j] = yv * zv * sigmoidf_(zv);
        ss += g[j] * g[j];
    }
    red[t] = ss;
    __syncthreads();
    for (int s = 64; s > 0; s >>= 1) {
        if (t < s) red[t] += red[t + s];
        __syncthreads();
    }
    const float scale = rsqrtf(red[0] / (float)D_INNER + 1e-5f);
#pragma unroll
    for (int j = 0; j < 3; ++j) {
        int d = t + j * 128;
        yb[(size_t)m * D_INNER + d] = f2bf(g[j] * scale * norm_w[d]);
    }
}

// ---------------- residual add + LayerNorm -> x (f32) and xb (bf16) -------
__global__ void k_res_ln(const float* __restrict__ ot, float* __restrict__ x,
                         u16* __restrict__ xb, const float* __restrict__ lnw,
                         const float* __restrict__ lnb) {
    __shared__ float s1[256], s2[256];
    const int m = blockIdx.x;
    const int t = threadIdx.x;     // 256 threads, first 192 active for data
    float v = 0.f;
    if (t < D_MODEL) v = ot[(size_t)m * D_MODEL + t] + x[(size_t)m * D_MODEL + t];
    s1[t] = v; s2[t] = v * v;
    __syncthreads();
    for (int s = 128; s > 0; s >>= 1) {
        if (t < s) { s1[t] += s1[t + s]; s2[t] += s2[t + s]; }
        __syncthreads();
    }
    const float mean = s1[0] / (float)D_MODEL;
    const float var  = s2[0] / (float)D_MODEL - mean * mean;
    if (t < D_MODEL) {
        float xn = (v - mean) * rsqrtf(var + 1e-5f) * lnw[t] + lnb[t];
        x[(size_t)m * D_MODEL + t]  = xn;
        xb[(size_t)m * D_MODEL + t] = f2bf(xn);
    }
}

// ---------------- output linear, out (B, 4, L) ----------------
__global__ void k_final(const float* __restrict__ x, const float* __restrict__ w,
                        const float* __restrict__ bias, float* __restrict__ out) {
    int i = blockIdx.x * blockDim.x + threadIdx.x;
    if (i >= MTOT * OUTPUT_DIM) return;
    int m = i / OUTPUT_DIM, o = i % OUTPUT_DIM;
    float acc = bias[o];
    for (int d = 0; d < D_MODEL; ++d)
        acc += x[(size_t)m * D_MODEL + d] * w[o * D_MODEL + d];
    int b = m / SEQLEN, l = m % SEQLEN;
    out[(size_t)(b * OUTPUT_DIM + o) * SEQLEN + l] = acc;
}

// ------------------------------------------------------------------
extern "C" void kernel_launch(void* const* d_in, const int* in_sizes, int n_in,
                              void* d_out, int out_size, void* d_ws, size_t ws_size,
                              hipStream_t stream) {
    const float* inp      = (const float*)d_in[0];
    const float* lin_in_w = (const float*)d_in[1];
    const float* lin_in_b = (const float*)d_in[2];
    const float* in_w     = (const float*)d_in[3];
    const float* conv_w   = (const float*)d_in[4];
    const float* conv_b   = (const float*)d_in[5];
    const float* dt_bias  = (const float*)d_in[6];
    const float* A_log    = (const float*)d_in[7];
    const float* Dp       = (const float*)d_in[8];
    const float* norm_w   = (const float*)d_in[9];
    const float* out_w    = (const float*)d_in[10];
    const float* ln_w     = (const float*)d_in[11];
    const float* ln_b     = (const float*)d_in[12];
    const float* lout_w   = (const float*)d_in[13];
    const float* lout_b   = (const float*)d_in[14];

    char* ws = (char*)d_ws;
    size_t off = 0;
    auto carve = [&](size_t bytes) -> char* {
        off = (off + 255) & ~(size_t)255;
        char* p = ws + off;
        off += bytes;
        return p;
    };
    float* x     = (float*)carve((size_t)MTOT * D_MODEL * 4);
    u16*   xb    = (u16*)  carve((size_t)MTOT * D_MODEL * 2);
    float* zxb   = (float*)carve((size_t)MTOT * NPAD * 4);
    float* xBC   = (float*)carve((size_t)MTOT * CONV_DIM * 4);
    float* dtb   = (float*)carve((size_t)MTOT * NHEADS * 4);
    float* ysb   = (float*)carve((size_t)MTOT * D_INNER * 4);
    u16*   yb    = (u16*)  carve((size_t)MTOT * D_INNER * 2);
    float* otb   = (float*)carve((size_t)MTOT * D_MODEL * 4);
    u16*   inwb  = (u16*)  carve((size_t)NLAYERS * NPAD * D_MODEL * 2);
    u16*   outwb = (u16*)  carve((size_t)NLAYERS * D_MODEL * D_INNER * 2);
    (void)ws_size; (void)in_sizes; (void)n_in; (void)out_size;

    k_pack_inw <<<(NLAYERS * NPAD * D_MODEL + 255) / 256, 256, 0, stream>>>(in_w, inwb);
    k_pack_outw<<<(NLAYERS * D_MODEL * D_INNER + 255) / 256, 256, 0, stream>>>(out_w, outwb);
    k_lin_in   <<<(MTOT * D_MODEL + 255) / 256, 256, 0, stream>>>(inp, lin_in_w, lin_in_b, x, xb);

    for (int i = 0; i < NLAYERS; ++i) {
        // in_proj: (M,192) x (960,192)^T -> (M,960)
        k_gemm_bf16<<<dim3(MTOT / 16 / GEMM_WAVES, NPAD / 64), 32 * GEMM_WAVES, 0, stream>>>(
            xb, inwb + (size_t)i * NPAD * D_MODEL, zxb, MTOT, NPAD, D_MODEL, NPAD);
        k_dt  <<<(MTOT * NHEADS + 255) / 256, 256, 0, stream>>>(zxb, dt_bias + i * NHEADS, dtb);
        k_conv<<<(MTOT * CONV_DIM + 255) / 256, 256, 0, stream>>>(
            zxb, conv_w + (size_t)i * CONV_DIM * D_CONV, conv_b + i * CONV_DIM, xBC);
        k_scan<<<dim3(BATCH, NHEADS), 256, 0, stream>>>(xBC, dtb, A_log + i * NHEADS, ysb);
        k_gate_rms<<<MTOT, 128, 0, stream>>>(ysb, xBC, zxb, Dp + i * NHEADS,
                                             norm_w + (size_t)i * D_INNER, yb);
        // out_proj: (M,384) x (192,384)^T -> (M,192)
        k_gemm_bf16<<<dim3(MTOT / 16 / GEMM_WAVES, D_MODEL / 64), 32 * GEMM_WAVES, 0, stream>>>(
            yb, outwb + (size_t)i * D_MODEL * D_INNER, otb, MTOT, D_MODEL, D_INNER, D_MODEL);
        k_res_ln<<<MTOT, 256, 0, stream>>>(otb, x, xb, ln_w + i * D_MODEL, ln_b + i * D_MODEL);
    }
    k_final<<<(MTOT * OUTPUT_DIM + 255) / 256, 256, 0, stream>>>(x, lout_w, lout_b, (float*)d_out);
}
